// Graph2Seq_85736137163255
// MI455X (gfx1250) — compile-verified, matrix operand order checked by reading, not run on hardware
//
#include <hip/hip_runtime.h>
#include <hip/hip_bf16.h>

// ---------------------------------------------------------------------------
// Graph2Seq (attention-model TSP-style encoder + pointer decoder) for gfx1250
// B=512, N=100, D=128, H=8, L=3, FF=512, DH=16, M = B*N = 51200 tokens
// GEMMs run on v_wmma_f32_16x16x32_f16 (f16 in, f32 accumulate).
// Each wave computes a 16x64 output strip (4 accumulators); the inner loop is
// software-pipelined: next K-step fragments are prefetched into a second
// register set while the current step's 4 WMMAs execute.
// ---------------------------------------------------------------------------

#define B_  512
#define N_  100
#define D_  128
#define H_  8
#define L_  3
#define FF_ 512
#define DH_ 16
#define M_  (B_ * N_)   // 51200 tokens

typedef _Float16 v16h  __attribute__((ext_vector_type(16)));
typedef _Float16 half8 __attribute__((ext_vector_type(8)));
typedef float    v8f   __attribute__((ext_vector_type(8)));

// ---------------- input projection: h = x @ W_in + b_in ----------------
__global__ void k_input_proj(const float* __restrict__ x, const float* __restrict__ Win,
                             const float* __restrict__ bin,
                             float* __restrict__ hf, _Float16* __restrict__ hh, int total) {
    int i = blockIdx.x * blockDim.x + threadIdx.x;
    if (i >= total) return;
    int tok = i >> 7, d = i & 127;
    float v = x[tok * 2 + 0] * Win[d] + x[tok * 2 + 1] * Win[D_ + d] + bin[d];
    hf[i] = v;
    hh[i] = (_Float16)v;
}

// ---------------- pack f32 weight [K,Nc] into WMMA B-fragment tiles ---------
// Tile (kt, nt): K rows kt*32..+31, cols nt*16..+15, nt-major tile order.
// Lane l owns col = nt*16+(l&15); 16 halves: k = kb..kb+7, kb+16..kb+23,
// kb = kt*32 + (l>>4)*8; stored contiguously (one 32B chunk per lane).
__global__ void k_pack_w(const float* __restrict__ W, _Float16* __restrict__ P,
                         int K, int Nc) {
    int tid = blockIdx.x * blockDim.x + threadIdx.x;
    int total = (K >> 5) * (Nc >> 4) * 32;
    if (tid >= total) return;
    int lane = tid & 31, tile = tid >> 5;
    int ktiles = K >> 5;
    int nt = tile / ktiles, kt = tile % ktiles;
    int col = nt * 16 + (lane & 15);
    int kb  = kt * 32 + (lane >> 4) * 8;
    _Float16* dst = P + (size_t)tile * 512 + lane * 16;
    #pragma unroll
    for (int j = 0; j < 8; ++j) dst[j]     = (_Float16)W[(size_t)(kb + j) * Nc + col];
    #pragma unroll
    for (int j = 0; j < 8; ++j) dst[8 + j] = (_Float16)W[(size_t)(kb + 16 + j) * Nc + col];
}

// ---------------- WMMA GEMM: out = act(A[M,K] @ Wp + bias) ------------------
// blockDim = 128 (4 waves). Wave w computes M-tile (mb*4 + w) x 64-wide
// N-strip (4 tiles, 4 accumulators). Software-pipelined fragment prefetch.
// Requires M % 64 == 0 and Nc % 64 == 0.
__global__ void k_gemm_wmma(const _Float16* __restrict__ A, const _Float16* __restrict__ Wp,
                            const float* __restrict__ bias,
                            float* __restrict__ outF, _Float16* __restrict__ outH,
                            int M, int K, int Nc, int relu) {
    int ngroups = Nc >> 6;                       // 64-wide N strips
    int blk = blockIdx.x;
    int mb  = blk / ngroups;                     // block of 4 M-tiles
    int ng  = blk % ngroups;
    int wave = threadIdx.x >> 5;
    int lane = threadIdx.x & 31;
    int mt   = mb * 4 + wave;
    int row0 = mt << 4;
    int khalf = lane >> 4;
    int kts  = K >> 5;

    v8f acc0 = {}, acc1 = {}, acc2 = {}, acc3 = {};

    const _Float16* Ab = A + (size_t)(row0 + (lane & 15)) * K + khalf * 8;
    const size_t strideB = (size_t)kts * 512;    // halves per nt stream
    const _Float16* Bg = Wp + (size_t)(ng * 4) * strideB + lane * 16;

    auto ldA = [&](int kt) -> v16h {
        half8 alo = *(const half8*)(Ab + kt * 32);
        half8 ahi = *(const half8*)(Ab + kt * 32 + 16);
        return __builtin_shufflevector(alo, ahi,
               0, 1, 2, 3, 4, 5, 6, 7, 8, 9, 10, 11, 12, 13, 14, 15);
    };

    // pipeline prologue: fragments for kt = 0
    v16h a  = ldA(0);
    v16h b0 = *(const v16h*)(Bg + 0 * strideB);
    v16h b1 = *(const v16h*)(Bg + 1 * strideB);
    v16h b2 = *(const v16h*)(Bg + 2 * strideB);
    v16h b3 = *(const v16h*)(Bg + 3 * strideB);

    for (int kt = 0; kt < kts - 1; ++kt) {
        // prefetch next K-step into a second register set
        v16h an = ldA(kt + 1);
        v16h c0 = *(const v16h*)(Bg + 0 * strideB + (kt + 1) * 512);
        v16h c1 = *(const v16h*)(Bg + 1 * strideB + (kt + 1) * 512);
        v16h c2 = *(const v16h*)(Bg + 2 * strideB + (kt + 1) * 512);
        v16h c3 = *(const v16h*)(Bg + 3 * strideB + (kt + 1) * 512);
        acc0 = __builtin_amdgcn_wmma_f32_16x16x32_f16(false, a, false, b0,
                                                      (short)0, acc0, false, false);
        acc1 = __builtin_amdgcn_wmma_f32_16x16x32_f16(false, a, false, b1,
                                                      (short)0, acc1, false, false);
        acc2 = __builtin_amdgcn_wmma_f32_16x16x32_f16(false, a, false, b2,
                                                      (short)0, acc2, false, false);
        acc3 = __builtin_amdgcn_wmma_f32_16x16x32_f16(false, a, false, b3,
                                                      (short)0, acc3, false, false);
        a = an; b0 = c0; b1 = c1; b2 = c2; b3 = c3;
    }
    // epilogue: last K-step
    acc0 = __builtin_amdgcn_wmma_f32_16x16x32_f16(false, a, false, b0,
                                                  (short)0, acc0, false, false);
    acc1 = __builtin_amdgcn_wmma_f32_16x16x32_f16(false, a, false, b1,
                                                  (short)0, acc1, false, false);
    acc2 = __builtin_amdgcn_wmma_f32_16x16x32_f16(false, a, false, b2,
                                                  (short)0, acc2, false, false);
    acc3 = __builtin_amdgcn_wmma_f32_16x16x32_f16(false, a, false, b3,
                                                  (short)0, acc3, false, false);

    v8f accs[4] = {acc0, acc1, acc2, acc3};
    #pragma unroll
    for (int i = 0; i < 4; ++i) {
        int col = (ng * 4 + i) * 16 + (lane & 15);
        float bv = bias ? bias[col] : 0.f;
        #pragma unroll
        for (int r = 0; r < 8; ++r) {
            int row = row0 + r + khalf * 8;
            float v = accs[i][r] + bv;
            if (relu) v = fmaxf(v, 0.f);
            size_t o = (size_t)row * Nc + col;
            if (outF) outF[o] = v;
            if (outH) outH[o] = (_Float16)v;
        }
    }
}

// ---------------- encoder attention (per batch,head; online softmax) --------
__global__ void k_attn_enc(const float* __restrict__ Q, const float* __restrict__ Kx,
                           const float* __restrict__ V, _Float16* __restrict__ O) {
    __shared__ float Ks[N_ * DH_];
    __shared__ float Vs[N_ * DH_];
    int bh = blockIdx.x;
    int b = bh / H_, h = bh % H_;
    int t = threadIdx.x;
    size_t base = (size_t)b * N_ * D_ + h * DH_;
    for (int i = t; i < N_ * DH_; i += blockDim.x) {
        int n = i >> 4, d = i & 15;
        Ks[i] = Kx[base + (size_t)n * D_ + d];
        Vs[i] = V [base + (size_t)n * D_ + d];
    }
    __syncthreads();
    if (t < N_) {
        float q[DH_], o[DH_];
        #pragma unroll
        for (int d = 0; d < DH_; ++d) { q[d] = Q[base + (size_t)t * D_ + d]; o[d] = 0.f; }
        float m = -1e30f, l = 0.f;
        for (int n = 0; n < N_; ++n) {
            float s = 0.f;
            #pragma unroll
            for (int d = 0; d < DH_; ++d) s += q[d] * Ks[n * DH_ + d];
            s *= 0.25f;                            // DH^-0.5
            float mn = fmaxf(m, s);
            float c  = __expf(m - mn);
            float p  = __expf(s - mn);
            l = l * c + p;
            #pragma unroll
            for (int d = 0; d < DH_; ++d) o[d] = o[d] * c + p * Vs[n * DH_ + d];
            m = mn;
        }
        float inv = 1.f / l;
        #pragma unroll
        for (int d = 0; d < DH_; ++d)
            O[base + (size_t)t * D_ + d] = (_Float16)(o[d] * inv);
    }
}

// ---------------- residual + LayerNorm (one wave32 per token) ---------------
__global__ void k_add_ln(float* __restrict__ h, const float* __restrict__ r,
                         const float* __restrict__ g, const float* __restrict__ bb,
                         _Float16* __restrict__ hh) {
    int wave = threadIdx.x >> 5, lane = threadIdx.x & 31;
    int tok = blockIdx.x * 8 + wave;                // blockDim = 256 = 8 waves
    if (tok >= M_) return;
    float x[4];
    float s = 0.f;
    #pragma unroll
    for (int j = 0; j < 4; ++j) {
        int d = j * 32 + lane;
        x[j] = h[(size_t)tok * D_ + d] + r[(size_t)tok * D_ + d];
        s += x[j];
    }
    for (int off = 16; off; off >>= 1) s += __shfl_xor(s, off, 32);
    float mean = s * (1.f / 128.f);
    float vs = 0.f;
    #pragma unroll
    for (int j = 0; j < 4; ++j) { float dd = x[j] - mean; vs += dd * dd; }
    for (int off = 16; off; off >>= 1) vs += __shfl_xor(vs, off, 32);
    float rstd = rsqrtf(vs * (1.f / 128.f) + 1e-5f);
    #pragma unroll
    for (int j = 0; j < 4; ++j) {
        int d = j * 32 + lane;
        float y = g[d] * (x[j] - mean) * rstd + bb[d];
        h[(size_t)tok * D_ + d]  = y;
        hh[(size_t)tok * D_ + d] = (_Float16)y;
    }
}

// ---------------- graph embedding: mean over N ------------------------------
__global__ void k_mean(const float* __restrict__ h, float* __restrict__ ge) {
    int i = blockIdx.x * blockDim.x + threadIdx.x;
    if (i >= B_ * D_) return;
    int b = i >> 7, d = i & 127;
    float s = 0.f;
    for (int n = 0; n < N_; ++n) s += h[((size_t)b * N_ + n) * D_ + d];
    ge[i] = s * (1.f / N_);
}

// ---------------- decoder state init ----------------------------------------
__global__ void k_dec_init(const float* __restrict__ t1, const float* __restrict__ tf,
                           float* __restrict__ second, float* __restrict__ third,
                           float* __restrict__ mask) {
    int i = blockIdx.x * blockDim.x + threadIdx.x;
    if (i < B_ * D_) { second[i] = t1[i & 127]; third[i] = tf[i & 127]; }
    if (i < B_ * N_) mask[i] = 0.f;
}

// ---------------- one greedy decode step (one block per batch element) ------
__global__ void k_dec_step(const float* __restrict__ nodes, const float* __restrict__ ge,
                           float* __restrict__ second, float* __restrict__ third,
                           float* __restrict__ mask,
                           const float* __restrict__ kh, const float* __restrict__ vh,
                           const float* __restrict__ k2,
                           const float* __restrict__ Wq, const float* __restrict__ Wo,
                           float* __restrict__ probs, float* __restrict__ tours, int step) {
    __shared__ float ctx[3 * D_];
    __shared__ float qv[D_];
    __shared__ float sc[H_][N_];
    __shared__ float gv[D_];
    __shared__ float go[D_];
    __shared__ float lg[N_];
    __shared__ int   sidx;

    int b = blockIdx.x, t = threadIdx.x;            // blockDim = 128
    ctx[t]          = ge[b * D_ + t];
    ctx[D_ + t]     = second[b * D_ + t];
    ctx[2 * D_ + t] = third[b * D_ + t];
    __syncthreads();

    // q = ctx @ dec_Wq  [384 x 128]
    float a = 0.f;
    for (int k = 0; k < 3 * D_; ++k) a += ctx[k] * Wq[k * D_ + t];
    qv[t] = a;
    __syncthreads();

    // masked scores over nodes, per head
    int h = t >> 4, j = t & 15;
    for (int n = j; n < N_; n += 16) {
        float s = 0.f;
        const float* kp = kh + ((size_t)b * N_ + n) * D_ + h * DH_;
        #pragma unroll
        for (int d = 0; d < DH_; ++d) s += qv[h * DH_ + d] * kp[d];
        s *= 0.25f;
        if (mask[b * N_ + n] > 0.f) s = -1e9f;
        sc[h][n] = s;
    }
    __syncthreads();

    // per-head softmax
    if (t < H_) {
        float m = -1e30f;
        for (int n = 0; n < N_; ++n) m = fmaxf(m, sc[t][n]);
        float sum = 0.f;
        for (int n = 0; n < N_; ++n) { float e = __expf(sc[t][n] - m); sc[t][n] = e; sum += e; }
        float inv = 1.f / sum;
        for (int n = 0; n < N_; ++n) sc[t][n] *= inv;
    }
    __syncthreads();

    // glimpse g = P @ v_h, then g @ dec_Wo
    float gacc = 0.f;
    for (int n = 0; n < N_; ++n) gacc += sc[t >> 4][n] * vh[((size_t)b * N_ + n) * D_ + t];
    gv[t] = gacc;
    __syncthreads();
    float oa = 0.f;
    for (int k = 0; k < D_; ++k) oa += gv[k] * Wo[k * D_ + t];
    go[t] = oa;
    __syncthreads();

    // pointer logits
    if (t < N_) {
        float s = 0.f;
        const float* kp = k2 + ((size_t)b * N_ + t) * D_;
        for (int d = 0; d < D_; ++d) s += go[d] * kp[d];
        s = 10.f * tanhf(s * 0.08838834764831845f);   // D^-0.5
        if (mask[b * N_ + t] > 0.f) s = -1e9f;
        lg[t] = s;
    }
    __syncthreads();

    // softmax + greedy argmax (argmax(prob) == argmax(logits), first max wins)
    if (t == 0) {
        float m = -1e30f; int bi = 0;
        for (int n = 0; n < N_; ++n) if (lg[n] > m) { m = lg[n]; bi = n; }
        float sum = 0.f;
        for (int n = 0; n < N_; ++n) { float e = __expf(lg[n] - m); lg[n] = e; sum += e; }
        float inv = 1.f / sum;
        for (int n = 0; n < N_; ++n) lg[n] *= inv;
        sidx = bi;
        tours[b * N_ + step] = (float)bi;
        mask[b * N_ + bi] = 1.f;
    }
    __syncthreads();
    if (t < N_) probs[((size_t)b * N_ + step) * N_ + t] = lg[t];

    int idx = sidx;
    float nv = nodes[((size_t)b * N_ + idx) * D_ + t];
    third[b * D_ + t] = nv;
    if (step == 0) second[b * D_ + t] = nv;
}

// ---------------------------------------------------------------------------
extern "C" void kernel_launch(void* const* d_in, const int* in_sizes, int n_in,
                              void* d_out, int out_size, void* d_ws, size_t ws_size,
                              hipStream_t stream) {
    (void)in_sizes; (void)n_in; (void)out_size; (void)ws_size;
    const float* x     = (const float*)d_in[0];
    const float* W_in  = (const float*)d_in[1];
    const float* b_in  = (const float*)d_in[2];
    const float* eWq   = (const float*)d_in[3];
    const float* eWk   = (const float*)d_in[4];
    const float* eWv   = (const float*)d_in[5];
    const float* eWo   = (const float*)d_in[6];
    const float* ln1g  = (const float*)d_in[7];
    const float* ln1b  = (const float*)d_in[8];
    const float* eW1   = (const float*)d_in[9];
    const float* eb1   = (const float*)d_in[10];
    const float* eW2   = (const float*)d_in[11];
    const float* eb2   = (const float*)d_in[12];
    const float* ln2g  = (const float*)d_in[13];
    const float* ln2b  = (const float*)d_in[14];
    const float* tok1  = (const float*)d_in[15];
    const float* tokf  = (const float*)d_in[16];
    const float* dWq   = (const float*)d_in[17];
    const float* dWk   = (const float*)d_in[18];
    const float* dWv   = (const float*)d_in[19];
    const float* dWo   = (const float*)d_in[20];
    const float* dWk2  = (const float*)d_in[21];
    float* out = (float*)d_out;
    float* probs = out;                    // [B,N,N]
    float* tours = out + (size_t)B_ * N_ * N_;

    char* ws = (char*)d_ws;
    size_t off = 0;
    auto alloc = [&](size_t bytes) -> void* {
        void* p = ws + off;
        off = (off + bytes + 255) & ~(size_t)255;
        return p;
    };

    float*    hf   = (float*)   alloc((size_t)M_ * D_ * 4);
    _Float16* hh   = (_Float16*)alloc((size_t)M_ * D_ * 2);
    float*    t0   = (float*)   alloc((size_t)M_ * D_ * 4);
    float*    qf   = (float*)   alloc((size_t)M_ * D_ * 4);   // later: k_h
    float*    kf   = (float*)   alloc((size_t)M_ * D_ * 4);   // later: v_h
    float*    vf   = (float*)   alloc((size_t)M_ * D_ * 4);   // later: k2
    _Float16* oh   = (_Float16*)alloc((size_t)M_ * D_ * 2);
    _Float16* mid  = (_Float16*)alloc((size_t)M_ * FF_ * 2);
    const size_t PK_DD = (size_t)(D_ / 32) * (D_ / 16) * 512;   // 16384 halves
    const size_t PK_DF = (size_t)(D_ / 32) * (FF_ / 16) * 512;  // 65536
    const size_t PK_FD = (size_t)(FF_ / 32) * (D_ / 16) * 512;  // 65536
    size_t packTotal = L_ * (4 * PK_DD + PK_DF + PK_FD) + 3 * PK_DD;
    _Float16* pw   = (_Float16*)alloc(packTotal * 2);
    float*    ge   = (float*)   alloc((size_t)B_ * D_ * 4);
    float*    sec  = (float*)   alloc((size_t)B_ * D_ * 4);
    float*    thr  = (float*)   alloc((size_t)B_ * D_ * 4);
    float*    msk  = (float*)   alloc((size_t)B_ * N_ * 4);

    // packed weight pointers
    _Float16 *pWq[L_], *pWk[L_], *pWv[L_], *pWo[L_], *pW1[L_], *pW2[L_];
    _Float16* pp = pw;
    for (int l = 0; l < L_; ++l) {
        pWq[l] = pp; pp += PK_DD;
        pWk[l] = pp; pp += PK_DD;
        pWv[l] = pp; pp += PK_DD;
        pWo[l] = pp; pp += PK_DD;
        pW1[l] = pp; pp += PK_DF;
        pW2[l] = pp; pp += PK_FD;
    }
    _Float16* pdWk  = pp; pp += PK_DD;
    _Float16* pdWv  = pp; pp += PK_DD;
    _Float16* pdWk2 = pp; pp += PK_DD;

    auto pack = [&](const float* W, _Float16* P, int K, int Nc) {
        int total = (K >> 5) * (Nc >> 4) * 32;
        k_pack_w<<<(total + 255) / 256, 256, 0, stream>>>(W, P, K, Nc);
    };
    auto gemm = [&](const _Float16* A, const _Float16* P, const float* bias,
                    float* oF, _Float16* oH, int M, int K, int Nc, int relu) {
        int blocks = (M >> 6) * (Nc >> 6);         // 4 M-tiles x 64-wide strip
        k_gemm_wmma<<<blocks, 128, 0, stream>>>(A, P, bias, oF, oH, M, K, Nc, relu);
    };

    // ---- weight packing (f32 -> f16 WMMA-B tiles) ----
    for (int l = 0; l < L_; ++l) {
        pack(eWq + (size_t)l * D_ * D_,  pWq[l], D_, D_);
        pack(eWk + (size_t)l * D_ * D_,  pWk[l], D_, D_);
        pack(eWv + (size_t)l * D_ * D_,  pWv[l], D_, D_);
        pack(eWo + (size_t)l * D_ * D_,  pWo[l], D_, D_);
        pack(eW1 + (size_t)l * D_ * FF_, pW1[l], D_, FF_);
        pack(eW2 + (size_t)l * FF_ * D_, pW2[l], FF_, D_);
    }
    pack(dWk,  pdWk,  D_, D_);
    pack(dWv,  pdWv,  D_, D_);
    pack(dWk2, pdWk2, D_, D_);

    // ---- input projection ----
    {
        int total = M_ * D_;
        k_input_proj<<<(total + 255) / 256, 256, 0, stream>>>(x, W_in, b_in, hf, hh, total);
    }

    // ---- encoder layers ----
    for (int l = 0; l < L_; ++l) {
        gemm(hh, pWq[l], nullptr, qf, nullptr, M_, D_, D_, 0);
        gemm(hh, pWk[l], nullptr, kf, nullptr, M_, D_, D_, 0);
        gemm(hh, pWv[l], nullptr, vf, nullptr, M_, D_, D_, 0);
        k_attn_enc<<<B_ * H_, 128, 0, stream>>>(qf, kf, vf, oh);
        gemm(oh, pWo[l], nullptr, t0, nullptr, M_, D_, D_, 0);
        k_add_ln<<<M_ / 8, 256, 0, stream>>>(hf, t0, ln1g + (size_t)l * D_,
                                             ln1b + (size_t)l * D_, hh);
        gemm(hh, pW1[l], eb1 + (size_t)l * FF_, nullptr, mid, M_, D_, FF_, 1);
        gemm(mid, pW2[l], eb2 + (size_t)l * D_, t0, nullptr, M_, FF_, D_, 0);
        k_add_ln<<<M_ / 8, 256, 0, stream>>>(hf, t0, ln2g + (size_t)l * D_,
                                             ln2b + (size_t)l * D_, hh);
    }

    // ---- decoder precompute ----
    k_mean<<<(B_ * D_ + 255) / 256, 256, 0, stream>>>(hf, ge);
    gemm(hh, pdWk,  nullptr, qf, nullptr, M_, D_, D_, 0);   // k_h
    gemm(hh, pdWv,  nullptr, kf, nullptr, M_, D_, D_, 0);   // v_h
    gemm(hh, pdWk2, nullptr, vf, nullptr, M_, D_, D_, 0);   // k2
    k_dec_init<<<(B_ * D_ + 255) / 256, 256, 0, stream>>>(tok1, tokf, sec, thr, msk);

    // ---- 100 sequential greedy decode steps ----
    for (int step = 0; step < N_; ++step) {
        k_dec_step<<<B_, 128, 0, stream>>>(hf, ge, sec, thr, msk,
                                           qf, kf, vf, dWq, dWo,
                                           probs, tours, step);
    }
}
